// MultiHeadAttention_22187801051747
// MI455X (gfx1250) — matrix operand order repaired and checked by reading, not compiled, not run on hardware
//
#include <hip/hip_runtime.h>

// ---------------------------------------------------------------------------
// MHA forward for MI455X (gfx1250, wave32, WMMA bf16 -> f32 accumulate)
//   B=4, T=2048, D=1024, H=16, Dh=64
// Pipeline:
//   1) cvt_f32_bf16:   x -> xbf                           [8192][1024] bf16
//   2) transpose_cvt:  w_qkv -> wqkvT [3072][1024] bf16 ; w_out -> woutT
//   3) qkv_gemm:       xbf @ w_qkv + b_qkv -> Q,K,V       [B,H,T,64] bf16
//                      (TDM double-buffered tiles, 8 WMMA / 64-K step)
//   4) attn (flash):   online-softmax(Q Kt / 8) V -> ctx  [B,T,1024] bf16
//   5) out_gemm:       ctx @ w_out + b_out -> out         f32
// Tile staging uses the Tensor Data Mover (tensor_load_to_lds + TENSORcnt)
// when the builtin exists; falls back to vector loads + ds stores otherwise.
// ---------------------------------------------------------------------------

typedef __attribute__((ext_vector_type(16))) __bf16        bf16x16;
typedef __attribute__((ext_vector_type(8)))  float         f32x8;
typedef __attribute__((ext_vector_type(4)))  unsigned int  u32x4;
typedef __attribute__((ext_vector_type(8)))  int           i32x8;
typedef __attribute__((ext_vector_type(4)))  int           i32x4;

union FragU {
  bf16x16 v;
  u32x4   u[2];
};

static __device__ __forceinline__ f32x8 zero8() {
  return (f32x8){0.f, 0.f, 0.f, 0.f, 0.f, 0.f, 0.f, 0.f};
}

// A fragment (M=16 x K=32 bf16) from row-major [m][k] LDS, stride in elems.
// ISA layout: lanes 0-15 -> row lane, K 0-7 & 16-23; lanes 16-31 -> row
// lane-16, K 8-15 & 24-31.  Two 16B ds reads per lane.
static __device__ __forceinline__ bf16x16 load_a_frag(const __bf16* base, int stride, int lane) {
  const int r  = lane & 15;
  const int hi = lane >> 4;
  const __bf16* p = base + r * stride + hi * 8;
  FragU f;
  f.u[0] = *(const u32x4*)(p);
  f.u[1] = *(const u32x4*)(p + 16);
  return f.v;
}

// B fragment (K=32 x N=16 bf16) from B^T row-major [n][k] LDS.
// ISA layout: lanes 0-15 -> col n = lane, K 0-7 & 16-23; lanes 16-31 -> K 8-15 & 24-31.
static __device__ __forceinline__ bf16x16 load_b_frag(const __bf16* base, int stride, int lane) {
  const int r  = lane & 15;
  const int hi = lane >> 4;
  const __bf16* p = base + r * stride + hi * 8;
  FragU f;
  f.u[0] = *(const u32x4*)(p);
  f.u[1] = *(const u32x4*)(p + 16);
  return f.v;
}

#define WMMA_BF16(a, b, c) \
  __builtin_amdgcn_wmma_f32_16x16x32_bf16(false, (a), false, (b), (short)0, (c), false, false)

// ---------------------------------------------------------------------------
// Tensor Data Mover tile staging (CDNA5).  D# per ISA cdna5 ch.8:
//   group0: count=1 | lds_addr | global_addr[56:0] | type=2
//   group1: data_size=1(2B), pad_enable, pad_interval=32dw, pad_amount=4dw,
//           tensor_dim0=64, tensor_dim1=rows, tile_dim0=64, tile_dim1=rows,
//           tensor_dim0_stride=rowStrideElems
// Pad: +16B after every 128B row -> LDS row stride 72 bf16 elements, which is
// exactly what the fragment loaders expect.
// ---------------------------------------------------------------------------
#if __has_builtin(__builtin_amdgcn_tensor_load_to_lds)
#define HAVE_TDM 1
#else
#define HAVE_TDM 0
#endif

static __device__ __forceinline__ unsigned lds_byte_off(const void* p) {
  // low 32 bits of a flat LDS address == workgroup LDS byte offset
  return (unsigned)(size_t)p;
}

#if HAVE_TDM
static __device__ __forceinline__ void tdm_load_tile64(
    unsigned lds_off, const __bf16* gsrc, unsigned rows, unsigned rowStrideElems) {
  unsigned long long ga = (unsigned long long)(size_t)gsrc;
  u32x4 g0;
  g0.x = 0x1u;                                  // count=1, user descriptor
  g0.y = lds_off;                               // lds_addr (bytes)
  g0.z = (unsigned)ga;                          // global_addr[31:0]
  g0.w = ((unsigned)(ga >> 32) & 0x01ffffffu)   // global_addr[56:32]
       | (2u << 30);                            // type=2 ("image")
  i32x8 g1;
  g1[0] = (int)((1u << 16)     // data_size: 2-byte elements
              | (1u << 20)     // pad_enable
              | (4u << 22)     // pad_interval code 4 -> every 32 DWORDs
              | (3u << 25));   // pad_amount  code 3 -> 4 DWORDs (16B)
  g1[1] = (int)(64u << 16);                 // tensor_dim0[15:0] = 64 @ bits 63:48
  g1[2] = (int)((rows & 0xffffu) << 16);    // tensor_dim0[31:16]=0 | tensor_dim1[15:0]
  g1[3] = (int)(64u << 16);                 // tensor_dim1[31:16]=0 | tile_dim0 = 64
  g1[4] = (int)(rows & 0xffffu);            // tile_dim1 = rows | tile_dim2 = 0
  g1[5] = (int)rowStrideElems;              // tensor_dim0_stride[31:0]
  g1[6] = 0;                                // stride[47:32]=0 | dim1_stride lo = 0
  g1[7] = 0;
  i32x4 z4 = (i32x4){0, 0, 0, 0};
#if __clang_major__ >= 23
  i32x8 z8 = (i32x8){0, 0, 0, 0, 0, 0, 0, 0};
  __builtin_amdgcn_tensor_load_to_lds(g0, g1, z4, z4, z8, 0);
#else
  __builtin_amdgcn_tensor_load_to_lds(g0, g1, z4, z4, 0);
#endif
}
#endif

static __device__ __forceinline__ void wait_tensor0() {
#if HAVE_TDM
#if __has_builtin(__builtin_amdgcn_s_wait_tensorcnt)
  __builtin_amdgcn_s_wait_tensorcnt(0);
#else
  asm volatile("s_wait_tensorcnt 0x0" ::: "memory");
#endif
#endif
}

// ---------------------------------------------------------------------------
// Prep kernels
// ---------------------------------------------------------------------------
__global__ void cvt_f32_bf16(const float* __restrict__ in, __bf16* __restrict__ out, int n) {
  int i = blockIdx.x * blockDim.x + threadIdx.x;
  if (i < n) out[i] = (__bf16)in[i];
}

// in[K][N] f32  ->  out[N][K] bf16
__global__ void transpose_cvt(const float* __restrict__ in, __bf16* __restrict__ out,
                              int K, int N) {
  long long i = (long long)blockIdx.x * blockDim.x + threadIdx.x;
  if (i < (long long)K * N) {
    int n  = (int)(i / K);
    int kk = (int)(i % K);
    out[i] = (__bf16)in[(long long)kk * N + n];
  }
}

// ---------------------------------------------------------------------------
// QKV projection GEMM: [8192][1024] x [1024][3072] + bias -> Q/K/V [B,H,T,64]
// Block tile 128x64, 8 waves (4x2) * 32x32 each, 64-wide K steps, TDM
// double-buffered LDS tiles.
// ---------------------------------------------------------------------------
__global__ __launch_bounds__(256) void qkv_gemm(
    const __bf16* __restrict__ A,    // xbf   [8192][1024]
    const __bf16* __restrict__ Bt,   // wqkvT [3072][1024]
    const float*  __restrict__ bias, // [3072]
    __bf16* __restrict__ Qo, __bf16* __restrict__ Ko, __bf16* __restrict__ Vo) {
  __shared__ __attribute__((aligned(128))) __bf16 As[2][128 * 72];
  __shared__ __attribute__((aligned(128))) __bf16 Bs[2][64 * 72];

  const int tid  = threadIdx.x;
  const int lane = tid & 31;
  const int w    = tid >> 5;
  const int bm = blockIdx.x & 63;   // 8192/128 = 64 M tiles
  const int bn = blockIdx.x >> 6;   // 3072/64  = 48 N tiles
  const int mBase = bm * 128, nBase = bn * 64;
  const int wm = (w >> 1) * 32, wn = (w & 1) * 32;
  const __bf16* Ag = A  + (size_t)mBase * 1024;
  const __bf16* Bg = Bt + (size_t)nBase * 1024;

  f32x8 acc[2][2];
  acc[0][0] = zero8(); acc[0][1] = zero8();
  acc[1][0] = zero8(); acc[1][1] = zero8();

  const int arow = tid >> 1, aseg = tid & 1;  // 128 rows x 2 chunks of 32 elems
  const int brow = tid >> 2, bseg = tid & 3;  // 64 rows x 4 chunks of 16 elems

  // ---- preload tile 0 --------------------------------------------------
#if HAVE_TDM
  if (w == 0) {
    tdm_load_tile64(lds_byte_off(&As[0][0]), Ag, 128u, 1024u);
    tdm_load_tile64(lds_byte_off(&Bs[0][0]), Bg, 64u, 1024u);
  }
#else
  {
    const u32x4* src = (const u32x4*)(Ag + (size_t)arow * 1024 + aseg * 32);
    u32x4* dst = (u32x4*)(&As[0][arow * 72 + aseg * 32]);
    dst[0] = src[0]; dst[1] = src[1]; dst[2] = src[2]; dst[3] = src[3];
  }
  {
    const u32x4* src = (const u32x4*)(Bg + (size_t)brow * 1024 + bseg * 16);
    u32x4* dst = (u32x4*)(&Bs[0][brow * 72 + bseg * 16]);
    dst[0] = src[0]; dst[1] = src[1];
  }
#endif
  wait_tensor0();
  __syncthreads();

  for (int k0 = 0; k0 < 1024; k0 += 64) {
    const int cur = (k0 >> 6) & 1, nxt = cur ^ 1;
    if (k0 + 64 < 1024) {  // stage next tile while computing current
#if HAVE_TDM
      if (w == 0) {
        tdm_load_tile64(lds_byte_off(&As[nxt][0]), Ag + k0 + 64, 128u, 1024u);
        tdm_load_tile64(lds_byte_off(&Bs[nxt][0]), Bg + k0 + 64, 64u, 1024u);
      }
#else
      {
        const u32x4* src = (const u32x4*)(Ag + (size_t)arow * 1024 + k0 + 64 + aseg * 32);
        u32x4* dst = (u32x4*)(&As[nxt][arow * 72 + aseg * 32]);
        dst[0] = src[0]; dst[1] = src[1]; dst[2] = src[2]; dst[3] = src[3];
      }
      {
        const u32x4* src = (const u32x4*)(Bg + (size_t)brow * 1024 + k0 + 64 + bseg * 16);
        u32x4* dst = (u32x4*)(&Bs[nxt][brow * 72 + bseg * 16]);
        dst[0] = src[0]; dst[1] = src[1];
      }
      if (k0 + 128 < 1024) {  // warm GL2/WGP$ one more tile ahead
        __builtin_prefetch(Ag + (size_t)arow * 1024 + k0 + 128, 0, 3);
        __builtin_prefetch(Bg + (size_t)brow * 1024 + k0 + 128, 0, 3);
      }
#endif
    }
    const __bf16* at = As[cur];
    const __bf16* bt = Bs[cur];
#pragma unroll
    for (int ks = 0; ks < 64; ks += 32) {
      bf16x16 af0 = load_a_frag(at + (size_t)wm * 72 + ks, 72, lane);
      bf16x16 af1 = load_a_frag(at + (size_t)(wm + 16) * 72 + ks, 72, lane);
      bf16x16 bf0 = load_b_frag(bt + (size_t)wn * 72 + ks, 72, lane);
      bf16x16 bf1 = load_b_frag(bt + (size_t)(wn + 16) * 72 + ks, 72, lane);
      acc[0][0] = WMMA_BF16(af0, bf0, acc[0][0]);
      acc[0][1] = WMMA_BF16(af0, bf1, acc[0][1]);
      acc[1][0] = WMMA_BF16(af1, bf0, acc[1][0]);
      acc[1][1] = WMMA_BF16(af1, bf1, acc[1][1]);
    }
    wait_tensor0();
    __syncthreads();
  }

  // Epilogue: +bias, scatter into [B,H,T,64] per q/k/v.
  const int nl = lane & 15, hi = lane >> 4;
  __bf16* bufs[3] = {Qo, Ko, Vo};
#pragma unroll
  for (int mi = 0; mi < 2; ++mi) {
#pragma unroll
    for (int ni = 0; ni < 2; ++ni) {
      const int gn = nBase + wn + ni * 16 + nl;
      const int which = gn >> 10, d = gn & 1023, h = d >> 6, dh = d & 63;
      const float bb = bias[gn];
      __bf16* dstb = bufs[which];
#pragma unroll
      for (int i = 0; i < 8; ++i) {
        const int gm = mBase + wm + mi * 16 + hi * 8 + i;
        const int b = gm >> 11, t = gm & 2047;
        dstb[((((size_t)b * 16 + h) * 2048) + t) * 64 + dh] = (__bf16)(acc[mi][ni][i] + bb);
      }
    }
  }
}

// ---------------------------------------------------------------------------
// Flash attention: grid = B*H * (T/128) = 1024 blocks of 256 threads.
// Each wave owns 16 queries; workgroup streams 64-key K/V blocks via LDS.
// K tile staged by TDM; V manually transposed into [d][key].
// ---------------------------------------------------------------------------
__global__ __launch_bounds__(256) void attn(
    const __bf16* __restrict__ Q, const __bf16* __restrict__ K,
    const __bf16* __restrict__ V, __bf16* __restrict__ ctx) {
  __shared__ __attribute__((aligned(128))) __bf16 Ks[64 * 72];    // [key][d]
  __shared__ __attribute__((aligned(128))) __bf16 Vts[64 * 72];   // [d][key]
  __shared__ __attribute__((aligned(128))) __bf16 Ps[8][16 * 40]; // per-wave P

  const int tid = threadIdx.x, lane = tid & 31, w = tid >> 5;
  const int nl = lane & 15, hi = lane >> 4;
  const int qtile = blockIdx.x & 15;   // 2048/128 = 16 query tiles
  const int bh = blockIdx.x >> 4;      // 0..63
  const int b = bh >> 4, h = bh & 15;
  const size_t base = (size_t)bh * 2048 * 64;
  const int q0 = qtile * 128 + w * 16;

  // Q A-fragments (Dh=64 -> two 16x32 fragments), loaded once from global.
  bf16x16 qa0, qa1;
  {
    const __bf16* qp = Q + base + (size_t)(q0 + nl) * 64;
    FragU f0, f1;
    f0.u[0] = *(const u32x4*)(qp + hi * 8);
    f0.u[1] = *(const u32x4*)(qp + hi * 8 + 16);
    f1.u[0] = *(const u32x4*)(qp + 32 + hi * 8);
    f1.u[1] = *(const u32x4*)(qp + 32 + hi * 8 + 16);
    qa0 = f0.v;
    qa1 = f1.v;
  }

  f32x8 O[4];
  O[0] = zero8(); O[1] = zero8(); O[2] = zero8(); O[3] = zero8();
  float mrow[8], lrow[8];
#pragma unroll
  for (int i = 0; i < 8; ++i) { mrow[i] = -3.0e38f; lrow[i] = 0.f; }

  const int ldr = tid >> 2, ldseg = tid & 3;  // 64 rows x 4 chunks of 16 elems

  for (int kb = 0; kb < 2048; kb += 64) {
    __syncthreads();
#if HAVE_TDM
    if (w == 0)  // K rows are contiguous [key][64]: stride 64, pad -> LDS 72
      tdm_load_tile64(lds_byte_off(Ks), K + base + (size_t)kb * 64, 64u, 64u);
#else
    {
      const u32x4* src = (const u32x4*)(K + base + (size_t)(kb + ldr) * 64 + ldseg * 16);
      u32x4* dst = (u32x4*)(Ks + ldr * 72 + ldseg * 16);
      dst[0] = src[0];
      dst[1] = src[1];
    }
#endif
    {  // V block, transposed to [d][key]
      const u32x4* src = (const u32x4*)(V + base + (size_t)(kb + ldr) * 64 + ldseg * 16);
      FragU f;
      f.u[0] = src[0];
      f.u[1] = src[1];
#pragma unroll
      for (int j = 0; j < 16; ++j) Vts[(ldseg * 16 + j) * 72 + ldr] = f.v[j];
    }
    wait_tensor0();
    __syncthreads();

#pragma unroll
    for (int kk = 0; kk < 64; kk += 32) {
      // Scores S (16 queries x 32 keys), f32, contraction over Dh=64.
      f32x8 S0 = zero8(), S1 = zero8();
      S0 = WMMA_BF16(qa0, load_b_frag(Ks + (size_t)(kk) * 72,           72, lane), S0);
      S0 = WMMA_BF16(qa1, load_b_frag(Ks + (size_t)(kk) * 72 + 32,      72, lane), S0);
      S1 = WMMA_BF16(qa0, load_b_frag(Ks + (size_t)(kk + 16) * 72,      72, lane), S1);
      S1 = WMMA_BF16(qa1, load_b_frag(Ks + (size_t)(kk + 16) * 72 + 32, 72, lane), S1);

      // Online softmax: scale 1/sqrt(64)=0.125, row stats over 32 keys.
      float rmax[8], p0a[8], p1a[8];
#pragma unroll
      for (int i = 0; i < 8; ++i) {
        float a = S0[i] * 0.125f, c = S1[i] * 0.125f;
        p0a[i] = a; p1a[i] = c;
        rmax[i] = fmaxf(a, c);
      }
#pragma unroll
      for (int off = 1; off < 16; off <<= 1) {
#pragma unroll
        for (int i = 0; i < 8; ++i)
          rmax[i] = fmaxf(rmax[i], __shfl_xor(rmax[i], off, 32));
      }
      float corr[8], rsum[8];
#pragma unroll
      for (int i = 0; i < 8; ++i) {
        float mnew = fmaxf(mrow[i], rmax[i]);
        corr[i] = __expf(mrow[i] - mnew);
        mrow[i] = mnew;
        float e0 = __expf(p0a[i] - mnew);
        float e1 = __expf(p1a[i] - mnew);
        p0a[i] = e0; p1a[i] = e1;
        rsum[i] = e0 + e1;
      }
#pragma unroll
      for (int off = 1; off < 16; off <<= 1) {
#pragma unroll
        for (int i = 0; i < 8; ++i) rsum[i] += __shfl_xor(rsum[i], off, 32);
      }
#pragma unroll
      for (int i = 0; i < 8; ++i) lrow[i] = lrow[i] * corr[i] + rsum[i];
#pragma unroll
      for (int j = 0; j < 4; ++j) {
#pragma unroll
        for (int i = 0; i < 8; ++i) O[j][i] *= corr[i];
      }

      // P: C-layout regs -> per-wave LDS (row-major [16][32+pad]) -> A frag.
      __bf16* ps = Ps[w];
#pragma unroll
      for (int i = 0; i < 8; ++i) {
        ps[(i + hi * 8) * 40 + nl]      = (__bf16)p0a[i];
        ps[(i + hi * 8) * 40 + 16 + nl] = (__bf16)p1a[i];
      }
      bf16x16 pa = load_a_frag(ps, 40, lane);

      // O += P @ V   (K-dim = 32 keys, N covers Dh=64 in 4 tiles)
#pragma unroll
      for (int j = 0; j < 4; ++j)
        O[j] = WMMA_BF16(pa, load_b_frag(Vts + (size_t)(j * 16) * 72 + kk, 72, lane), O[j]);
    }
  }

  // Normalize and write ctx[B][T][H*64] bf16.
#pragma unroll
  for (int j = 0; j < 4; ++j) {
#pragma unroll
    for (int i = 0; i < 8; ++i) {
      const int m = q0 + hi * 8 + i;
      const float val = O[j][i] / lrow[i];
      ctx[((size_t)b * 2048 + m) * 1024 + h * 64 + j * 16 + nl] = (__bf16)val;
    }
  }
}

// ---------------------------------------------------------------------------
// Output projection: ctx [8192][1024] x woutT [1024][1024] + b_out -> f32 out
// ---------------------------------------------------------------------------
__global__ __launch_bounds__(256) void out_gemm(
    const __bf16* __restrict__ A,    // ctx   [8192][1024]
    const __bf16* __restrict__ Bt,   // woutT [1024][1024]
    const float*  __restrict__ bias, // [1024]
    float* __restrict__ Out) {       // [8192][1024] f32
  __shared__ __attribute__((aligned(128))) __bf16 As[2][128 * 72];
  __shared__ __attribute__((aligned(128))) __bf16 Bs[2][64 * 72];

  const int tid  = threadIdx.x;
  const int lane = tid & 31;
  const int w    = tid >> 5;
  const int bm = blockIdx.x & 63;   // 64 M tiles
  const int bn = blockIdx.x >> 6;   // 16 N tiles
  const int mBase = bm * 128, nBase = bn * 64;
  const int wm = (w >> 1) * 32, wn = (w & 1) * 32;
  const __bf16* Ag = A  + (size_t)mBase * 1024;
  const __bf16* Bg = Bt + (size_t)nBase * 1024;

  f32x8 acc[2][2];
  acc[0][0] = zero8(); acc[0][1] = zero8();
  acc[1][0] = zero8(); acc[1][1] = zero8();

  const int arow = tid >> 1, aseg = tid & 1;
  const int brow = tid >> 2, bseg = tid & 3;

#if HAVE_TDM
  if (w == 0) {
    tdm_load_tile64(lds_byte_off(&As[0][0]), Ag, 128u, 1024u);
    tdm_load_tile64(lds_byte_off(&Bs[0][0]), Bg, 64u, 1024u);
  }
#else
  {
    const u32x4* src = (const u32x4*)(Ag + (size_t)arow * 1024 + aseg * 32);
    u32x4* dst = (u32x4*)(&As[0][arow * 72 + aseg * 32]);
    dst[0] = src[0]; dst[1] = src[1]; dst[2] = src[2]; dst[3] = src[3];
  }
  {
    const u32x4* src = (const u32x4*)(Bg + (size_t)brow * 1024 + bseg * 16);
    u32x4* dst = (u32x4*)(&Bs[0][brow * 72 + bseg * 16]);
    dst[0] = src[0]; dst[1] = src[1];
  }
#endif
  wait_tensor0();
  __syncthreads();

  for (int k0 = 0; k0 < 1024; k0 += 64) {
    const int cur = (k0 >> 6) & 1, nxt = cur ^ 1;
    if (k0 + 64 < 1024) {
#if HAVE_TDM
      if (w == 0) {
        tdm_load_tile64(lds_byte_off(&As[nxt][0]), Ag + k0 + 64, 128u, 1024u);
        tdm_load_tile64(lds_byte_off(&Bs[nxt][0]), Bg + k0 + 64, 64u, 1024u);
      }
#else
      {
        const u32x4* src = (const u32x4*)(Ag + (size_t)arow * 1024 + k0 + 64 + aseg * 32);
        u32x4* dst = (u32x4*)(&As[nxt][arow * 72 + aseg * 32]);
        dst[0] = src[0]; dst[1] = src[1]; dst[2] = src[2]; dst[3] = src[3];
      }
      {
        const u32x4* src = (const u32x4*)(Bg + (size_t)brow * 1024 + k0 + 64 + bseg * 16);
        u32x4* dst = (u32x4*)(&Bs[nxt][brow * 72 + bseg * 16]);
        dst[0] = src[0]; dst[1] = src[1];
      }
      if (k0 + 128 < 1024) {
        __builtin_prefetch(Ag + (size_t)arow * 1024 + k0 + 128, 0, 3);
        __builtin_prefetch(Bg + (size_t)brow * 1024 + k0 + 128, 0, 3);
      }
#endif
    }
    const __bf16* at = As[cur];
    const __bf16* bt = Bs[cur];
#pragma unroll
    for (int ks = 0; ks < 64; ks += 32) {
      bf16x16 af0 = load_a_frag(at + (size_t)wm * 72 + ks, 72, lane);
      bf16x16 af1 = load_a_frag(at + (size_t)(wm + 16) * 72 + ks, 72, lane);
      bf16x16 bf0 = load_b_frag(bt + (size_t)wn * 72 + ks, 72, lane);
      bf16x16 bf1 = load_b_frag(bt + (size_t)(wn + 16) * 72 + ks, 72, lane);
      acc[0][0] = WMMA_BF16(af0, bf0, acc[0][0]);
      acc[0][1] = WMMA_BF16(af0, bf1, acc[0][1]);
      acc[1][0] = WMMA_BF16(af1, bf0, acc[1][0]);
      acc[1][1] = WMMA_BF16(af1, bf1, acc[1][1]);
    }
    wait_tensor0();
    __syncthreads();
  }

  const int nl = lane & 15, hi = lane >> 4;
#pragma unroll
  for (int mi = 0; mi < 2; ++mi) {
#pragma unroll
    for (int ni = 0; ni < 2; ++ni) {
      const int gn = nBase + wn + ni * 16 + nl;
      const float bb = bias[gn];
#pragma unroll
      for (int i = 0; i < 8; ++i) {
        const int gm = mBase + wm + mi * 16 + hi * 8 + i;
        Out[(size_t)gm * 1024 + gn] = acc[mi][ni][i] + bb;
      }
    }
  }
}

// ---------------------------------------------------------------------------
// Host launch
// ---------------------------------------------------------------------------
extern "C" void kernel_launch(void* const* d_in, const int* in_sizes, int n_in,
                              void* d_out, int out_size, void* d_ws, size_t ws_size,
                              hipStream_t stream) {
  (void)in_sizes; (void)n_in; (void)out_size; (void)ws_size;

  const float* x     = (const float*)d_in[0];  // [4,2048,1024]
  const float* w_qkv = (const float*)d_in[1];  // [1024,3072]
  const float* b_qkv = (const float*)d_in[2];  // [3072]
  const float* w_out = (const float*)d_in[3];  // [1024,1024]
  const float* b_out = (const float*)d_in[4];  // [1024]
  float* out = (float*)d_out;                  // [4,2048,1024]

  char* ws = (char*)d_ws;
  const size_t OFF_XBF   = 0;                          // 16 MiB
  const size_t OFF_WQKVT = OFF_XBF   + 16777216;       //  6 MiB
  const size_t OFF_WOUTT = OFF_WQKVT + 6291456;        //  2 MiB
  const size_t OFF_Q     = OFF_WOUTT + 2097152;        // 16 MiB
  const size_t OFF_K     = OFF_Q     + 16777216;       // 16 MiB
  const size_t OFF_V     = OFF_K     + 16777216;       // 16 MiB
  const size_t OFF_CTX   = OFF_V     + 16777216;       // 16 MiB (total ~88 MiB)

  __bf16* xbf   = (__bf16*)(ws + OFF_XBF);
  __bf16* wqkvT = (__bf16*)(ws + OFF_WQKVT);
  __bf16* woutT = (__bf16*)(ws + OFF_WOUTT);
  __bf16* Qb    = (__bf16*)(ws + OFF_Q);
  __bf16* Kb    = (__bf16*)(ws + OFF_K);
  __bf16* Vb    = (__bf16*)(ws + OFF_V);
  __bf16* ctx   = (__bf16*)(ws + OFF_CTX);

  cvt_f32_bf16<<<32768, 256, 0, stream>>>(x, xbf, 8388608);
  transpose_cvt<<<12288, 256, 0, stream>>>(w_qkv, wqkvT, 1024, 3072);
  transpose_cvt<<<4096, 256, 0, stream>>>(w_out, woutT, 1024, 1024);
  qkv_gemm<<<3072, 256, 0, stream>>>(xbf, wqkvT, b_qkv, Qb, Kb, Vb);
  attn<<<1024, 256, 0, stream>>>(Qb, Kb, Vb, ctx);
  out_gemm<<<1024, 256, 0, stream>>>(ctx, woutT, b_out, out);
}
